// MultiScaleConvLayer_60498909331488
// MI455X (gfx1250) — matrix operand order repaired and checked by reading, not compiled, not run on hardware
//
#include <hip/hip_runtime.h>
#include <math.h>

// ---------------- problem constants ----------------
#define NATOMS 100000
#define MNBR   12
#define AFEA   64
#define FFEA   41
#define KPAD   192                 // 2A+F = 169 padded to 6 x K32 chunks
#define NF_PAD 48
#define ROWS1  (NATOMS*MNBR)       // 1,200,000  (divisible by 16)
#define TILES1 (ROWS1/16)          // 75,000
#define TILESF (NATOMS/16)         // 6,250
#define NB1    600                 // pass1 blocks (fixed -> deterministic reduction)
#define NB2    512                 // pass2 blocks
#define BN_EPS 1e-5f

typedef unsigned short ushort_t;
typedef __attribute__((ext_vector_type(16))) __bf16 v16bf;
typedef __attribute__((ext_vector_type(2)))  __bf16 v2bf;
typedef __attribute__((ext_vector_type(8)))  float  v8f;

union Frag { v16bf v; uint4 q[2]; };

// ---------------- helpers (native hardware bf16 converts) ----------------
__device__ __forceinline__ ushort_t f2bf(float f) {
  union { __bf16 b; ushort_t u; } c;
  c.b = (__bf16)f;                    // v_cvt bf16 (RNE) on gfx1250
  return c.u;
}
__device__ __forceinline__ float bf2f(ushort_t h) {
  return __uint_as_float(((unsigned)h) << 16);
}
__device__ __forceinline__ unsigned pk2(float a, float b) {
  union { v2bf v; unsigned u; } c;
  c.v[0] = (__bf16)a;
  c.v[1] = (__bf16)b;                 // packed convert pair
  return c.u;
}
__device__ __forceinline__ float softplusf(float x) {
  return (x > 20.f) ? x : log1pf(expf(x));
}
__device__ __forceinline__ float sigmoidf(float x) {
  return 1.f / (1.f + expf(-x));
}

// ---------------- weight prep: transpose + bf16 + zero-pad ----------------
// WT1: [3][128 cols][192 K], WTs: [2][48 cols][64 K], WTf: [64 cols][192 K]
__global__ void prep_kernel(const float* __restrict__ W_full,
                            const float* __restrict__ W_scale,
                            const float* __restrict__ W_fuse,
                            ushort_t* __restrict__ WT1,
                            ushort_t* __restrict__ WTs,
                            ushort_t* __restrict__ WTf) {
  int t = blockIdx.x * 256 + threadIdx.x;
  const int n1 = 3 * 128 * 192, n2 = 2 * 48 * 64, n3 = 64 * 192;
  if (t < n1) {
    int sc = t / (128 * 192);
    int rem = t - sc * 128 * 192;
    int col = rem / 192, k = rem % 192;
    WT1[t] = (k < 169) ? f2bf(W_full[((size_t)sc * 169 + k) * 128 + col]) : (ushort_t)0;
  } else if (t < n1 + n2) {
    int u = t - n1;
    int s = u / (48 * 64);
    int rem = u - s * 48 * 64;
    int col = rem / 64, k = rem % 64;
    WTs[u] = (k < FFEA && col < FFEA)
                 ? f2bf(W_scale[((size_t)s * FFEA + k) * FFEA + col]) : (ushort_t)0;
  } else if (t < n1 + n2 + n3) {
    int u = t - n1 - n2;
    int col = u / 192, k = u % 192;
    WTf[u] = f2bf(W_fuse[(size_t)k * 64 + col]);
  }
}

// ---------------- scale 0: copy nbr_fea -> padded bf16 ----------------
__global__ void nf0_kernel(const float* __restrict__ nbr, ushort_t* __restrict__ nf16) {
  long e = (long)blockIdx.x * 256 + threadIdx.x;
  if (e >= (long)ROWS1 * NF_PAD) return;
  int g = (int)(e / NF_PAD);
  int c = (int)(e - (long)g * NF_PAD);
  nf16[e] = (c < FFEA) ? f2bf(nbr[(size_t)g * FFEA + c]) : (ushort_t)0;
}

// ---------------- scales 1,2: nf = nbr_fea @ W_scale + b  (WMMA) ----------------
__global__ void nfgemm_kernel(const float* __restrict__ nbr,
                              const ushort_t* __restrict__ WTs,   // [48][64]
                              const float* __restrict__ bsc,      // [41]
                              ushort_t* __restrict__ nf16) {
  __shared__ __align__(16) ushort_t Ab[4][16][64];
  int tid = threadIdx.x, wave = tid >> 5, lane = tid & 31;
  int r = lane & 15, h = lane >> 4, col0 = lane & 15;
  int koA = h ? 16 : 0, koB = h ? 16 : 0;

  for (int tile = blockIdx.x * 4 + wave; tile < TILES1; tile += gridDim.x * 4) {
    int g = tile * 16 + r;
    const float* row = nbr + (size_t)g * FFEA;
    for (int c = 0; c < 32; ++c) {
      int col = h * 32 + c;
      Ab[wave][r][col] = (col < FFEA) ? f2bf(row[col]) : (ushort_t)0;
    }
    asm volatile("s_wait_dscnt 0x0" ::: "memory");   // cross-lane LDS visibility

    for (int nt = 0; nt < 3; ++nt) {
      int col = nt * 16 + col0;
      float b = (col < FFEA) ? bsc[col] : 0.f;
      v8f acc; for (int v = 0; v < 8; ++v) acc[v] = b;
      const ushort_t* wp = WTs + (size_t)col * 64;
      for (int kc = 0; kc < 2; ++kc) {
        int kb = kc * 32;
        Frag a, bw;
        a.q[0]  = *(const uint4*)&Ab[wave][r][kb + koA];
        a.q[1]  = *(const uint4*)&Ab[wave][r][kb + koA + 8];
        bw.q[0] = *(const uint4*)(wp + kb + koB);
        bw.q[1] = *(const uint4*)(wp + kb + koB + 8);
        acc = __builtin_amdgcn_wmma_f32_16x16x32_bf16(false, a.v, false, bw.v,
                                                      (short)0, acc, false, false);
      }
      int rb = h ? 8 : 0;
      for (int v = 0; v < 8; ++v) {
        ushort_t o = (col < FFEA) ? f2bf(acc[v]) : (ushort_t)0;  // keep pad finite
        nf16[(size_t)(tile * 16 + rb + v) * NF_PAD + col] = o;
      }
    }
  }
}

// ---------------- pass 1: gather+concat -> LDS bf16 A-tile -> WMMA z, BN1 stats ----------------
__global__ void pass1_kernel(const float* __restrict__ atom,
                             const int* __restrict__ idx,
                             const ushort_t* __restrict__ nf16,
                             const ushort_t* __restrict__ WT,    // [128][192]
                             const float* __restrict__ bias,     // [128]
                             ushort_t* __restrict__ z16,
                             float* __restrict__ p1) {
  __shared__ __align__(16) ushort_t Ab[4][16][KPAD];
  __shared__ float Ssum[128], Sssq[128];
  int tid = threadIdx.x;
  if (tid < 128) { Ssum[tid] = 0.f; Sssq[tid] = 0.f; }
  __syncthreads();

  int wave = tid >> 5, lane = tid & 31;
  int r = lane & 15, h = lane >> 4, col0 = lane & 15;
  int koA = h ? 16 : 0, koB = h ? 16 : 0;
  const int step = gridDim.x * 4;

  for (int tile = blockIdx.x * 4 + wave; tile < TILES1; tile += step) {
    int g = tile * 16 + r;
    int n = g / MNBR;
    int m = g - n * MNBR;
    int j = idx[n * MNBR + m];
    // prefetch next grid-stride tile's streamed operand (global_prefetch_b8)
    if (tile + step < TILES1)
      __builtin_prefetch(nf16 + (size_t)(tile + step) * 16 * NF_PAD + (size_t)r * NF_PAD, 0, 1);

    ushort_t (*A)[KPAD] = Ab[wave];
    if (h == 0) {                      // cols [0,96): self 64 + nbr 0..31
      const float4* self = (const float4*)(atom + (size_t)n * AFEA);
      for (int c4 = 0; c4 < 16; ++c4) {
        float4 v = self[c4];
        uint2 p; p.x = pk2(v.x, v.y); p.y = pk2(v.z, v.w);
        *(uint2*)&A[r][c4 * 4] = p;
      }
      const float4* nb = (const float4*)(atom + (size_t)j * AFEA);
      for (int c4 = 0; c4 < 8; ++c4) {
        float4 v = nb[c4];
        uint2 p; p.x = pk2(v.x, v.y); p.y = pk2(v.z, v.w);
        *(uint2*)&A[r][64 + c4 * 4] = p;
      }
    } else {                           // cols [96,192): nbr 32..63 + nf 48 + pad 16
      const float4* nb = (const float4*)(atom + (size_t)j * AFEA + 32);
      for (int c4 = 0; c4 < 8; ++c4) {
        float4 v = nb[c4];
        uint2 p; p.x = pk2(v.x, v.y); p.y = pk2(v.z, v.w);
        *(uint2*)&A[r][96 + c4 * 4] = p;
      }
      const uint2* nfr = (const uint2*)(nf16 + (size_t)g * NF_PAD);
      for (int k = 0; k < 12; ++k) *(uint2*)&A[r][128 + k * 4] = nfr[k];
      uint2 z; z.x = 0u; z.y = 0u;
      for (int k = 0; k < 4; ++k) *(uint2*)&A[r][176 + k * 4] = z;
    }
    asm volatile("s_wait_dscnt 0x0" ::: "memory");

    for (int nt = 0; nt < 8; ++nt) {
      int col = nt * 16 + col0;
      float b = bias[col];
      v8f acc; for (int v = 0; v < 8; ++v) acc[v] = b;
      const ushort_t* wp = WT + (size_t)col * KPAD;
      for (int kc = 0; kc < 6; ++kc) {
        int kb = kc * 32;
        Frag a, bw;
        a.q[0]  = *(const uint4*)&A[r][kb + koA];
        a.q[1]  = *(const uint4*)&A[r][kb + koA + 8];
        bw.q[0] = *(const uint4*)(wp + kb + koB);
        bw.q[1] = *(const uint4*)(wp + kb + koB + 8);
        acc = __builtin_amdgcn_wmma_f32_16x16x32_bf16(false, a.v, false, bw.v,
                                                      (short)0, acc, false, false);
      }
      int rb = h ? 8 : 0;
      float s = 0.f, q = 0.f;
      for (int v = 0; v < 8; ++v) {
        float zv = acc[v];
        z16[(size_t)(tile * 16 + rb + v) * 128 + col] = f2bf(zv);
        s += zv; q += zv * zv;
      }
      s += __shfl_down(s, 16, 32);     // lane l and l+16 share output column
      q += __shfl_down(q, 16, 32);
      if (h == 0) { atomicAdd(&Ssum[col], s); atomicAdd(&Sssq[col], q); }
    }
  }
  __syncthreads();
  if (tid < 128) {                     // fixed per-block partials -> deterministic tree
    p1[(size_t)blockIdx.x * 256 + tid]       = Ssum[tid];
    p1[(size_t)blockIdx.x * 256 + 128 + tid] = Sssq[tid];
  }
}

// ---------------- BN finalizers (fixed-order, deterministic) ----------------
__global__ void reduce1_kernel(const float* __restrict__ p1, const float* __restrict__ g1,
                               const float* __restrict__ bt1, float* __restrict__ bn1) {
  int t = threadIdx.x;                 // 128
  float S = 0.f, Q = 0.f;
  for (int b = 0; b < NB1; ++b) { S += p1[b * 256 + t]; Q += p1[b * 256 + 128 + t]; }
  const float cnt = (float)ROWS1;
  float mu = S / cnt;
  float var = Q / cnt - mu * mu;
  float sc = g1[t] * rsqrtf(var + BN_EPS);
  bn1[t] = sc;
  bn1[128 + t] = bt1[t] - mu * sc;
}
__global__ void reduce2_kernel(const float* __restrict__ p2, const float* __restrict__ g2,
                               const float* __restrict__ bt2, float* __restrict__ bn2) {
  int t = threadIdx.x;                 // 64
  float S = 0.f, Q = 0.f;
  for (int b = 0; b < NB2; ++b) { S += p2[b * 128 + t]; Q += p2[b * 128 + 64 + t]; }
  const float cnt = (float)NATOMS;
  float mu = S / cnt;
  float var = Q / cnt - mu * mu;
  float sc = g2[t] * rsqrtf(var + BN_EPS);
  bn2[t] = sc;
  bn2[64 + t] = bt2[t] - mu * sc;
}

// ---------------- pass 2: normalize z, sigmoid*softplus, sum over M, BN2 stats ----------------
__global__ void pass2_kernel(const ushort_t* __restrict__ z16,
                             const float* __restrict__ bn1,   // [0,128) scale [128,256) shift
                             float* __restrict__ summed,
                             float* __restrict__ p2) {
  __shared__ float S[64], Q[64];
  int tid = threadIdx.x;
  if (tid < 64) { S[tid] = 0.f; Q[tid] = 0.f; }
  __syncthreads();
  const long total = (long)NATOMS * AFEA;
  for (long e = (long)blockIdx.x * 256 + tid; e < total; e += (long)gridDim.x * 256) {
    int n = (int)(e >> 6), a = (int)(e & 63);
    float scf = bn1[a],      shf = bn1[128 + a];
    float scc = bn1[64 + a], shc = bn1[192 + a];
    const ushort_t* zr = z16 + (size_t)n * MNBR * 128;
    float s = 0.f;
    for (int m = 0; m < MNBR; ++m) {
      float zf = bf2f(zr[m * 128 + a])      * scf + shf;
      float zc = bf2f(zr[m * 128 + 64 + a]) * scc + shc;
      s += sigmoidf(zf) * softplusf(zc);
    }
    summed[e] = s;
    atomicAdd(&S[a], s);
    atomicAdd(&Q[a], s * s);
  }
  __syncthreads();
  if (tid < 64) {
    p2[(size_t)blockIdx.x * 128 + tid]      = S[tid];
    p2[(size_t)blockIdx.x * 128 + 64 + tid] = Q[tid];
  }
}

// ---------------- pass 3: softplus(atom + BN2(summed)) -> bf16 cat slice ----------------
__global__ void pass3_kernel(const float* __restrict__ atom, const float* __restrict__ summed,
                             const float* __restrict__ bn2, ushort_t* __restrict__ cat16, int si) {
  long e = (long)blockIdx.x * 256 + threadIdx.x;
  if (e >= (long)NATOMS * AFEA) return;
  int n = (int)(e >> 6), a = (int)(e & 63);
  float v = atom[e] + summed[e] * bn2[a] + bn2[64 + a];
  cat16[(size_t)n * 192 + si * 64 + a] = f2bf(softplusf(v));
}

// ---------------- fuse: relu(cat @ W_fuse + b)  (WMMA, A direct from global) ----------------
__global__ void fuse_kernel(const ushort_t* __restrict__ cat16,  // [N][192] bf16
                            const ushort_t* __restrict__ WTf,    // [64][192]
                            const float* __restrict__ bfuse,
                            float* __restrict__ out) {
  int tid = threadIdx.x, wave = tid >> 5, lane = tid & 31;
  int r = lane & 15, h = lane >> 4, col0 = lane & 15;
  int koA = h ? 16 : 0, koB = h ? 16 : 0;
  for (int tile = blockIdx.x * 4 + wave; tile < TILESF; tile += gridDim.x * 4) {
    const ushort_t* arow = cat16 + (size_t)(tile * 16 + r) * 192;
    for (int nt = 0; nt < 4; ++nt) {
      int col = nt * 16 + col0;
      float b = bfuse[col];
      v8f acc; for (int v = 0; v < 8; ++v) acc[v] = b;
      const ushort_t* wp = WTf + (size_t)col * 192;
      for (int kc = 0; kc < 6; ++kc) {
        int kb = kc * 32;
        Frag a, bw;
        a.q[0]  = *(const uint4*)(arow + kb + koA);
        a.q[1]  = *(const uint4*)(arow + kb + koA + 8);
        bw.q[0] = *(const uint4*)(wp + kb + koB);
        bw.q[1] = *(const uint4*)(wp + kb + koB + 8);
        acc = __builtin_amdgcn_wmma_f32_16x16x32_bf16(false, a.v, false, bw.v,
                                                      (short)0, acc, false, false);
      }
      int rb = h ? 8 : 0;
      for (int v = 0; v < 8; ++v)
        out[(size_t)(tile * 16 + rb + v) * 64 + col] = fmaxf(acc[v], 0.f);
    }
  }
}

// ---------------- launcher ----------------
extern "C" void kernel_launch(void* const* d_in, const int* in_sizes, int n_in,
                              void* d_out, int out_size, void* d_ws, size_t ws_size,
                              hipStream_t stream) {
  const float* atom    = (const float*)d_in[0];
  const float* nbr     = (const float*)d_in[1];
  const int*   idx     = (const int*)d_in[2];
  const float* W_full  = (const float*)d_in[3];
  const float* b_full  = (const float*)d_in[4];
  const float* g1      = (const float*)d_in[5];
  const float* bt1     = (const float*)d_in[6];
  const float* g2      = (const float*)d_in[7];
  const float* bt2     = (const float*)d_in[8];
  const float* W_scale = (const float*)d_in[9];
  const float* b_scale = (const float*)d_in[10];
  const float* W_fuse  = (const float*)d_in[11];
  const float* b_fuse  = (const float*)d_in[12];
  float* out = (float*)d_out;

  char* ws = (char*)d_ws;
  size_t off = 0;
  auto alloc = [&](size_t bytes) -> char* {
    char* p = ws + off;
    off = (off + bytes + 255) & ~(size_t)255;
    return p;
  };
  ushort_t* nf16  = (ushort_t*)alloc((size_t)ROWS1 * NF_PAD * 2);   // 115.2 MB
  ushort_t* z16   = (ushort_t*)alloc((size_t)ROWS1 * 128 * 2);      // 307.2 MB
  ushort_t* cat16 = (ushort_t*)alloc((size_t)NATOMS * 192 * 2);     //  38.4 MB
  float*    summed= (float*)   alloc((size_t)NATOMS * 64 * 4);      //  25.6 MB
  ushort_t* WT1   = (ushort_t*)alloc((size_t)3 * 128 * 192 * 2);
  ushort_t* WTs   = (ushort_t*)alloc((size_t)2 * 48 * 64 * 2);
  ushort_t* WTf   = (ushort_t*)alloc((size_t)64 * 192 * 2);
  float*    bn1   = (float*)   alloc(256 * 4);
  float*    bn2   = (float*)   alloc(128 * 4);
  float*    p1    = (float*)   alloc((size_t)NB1 * 256 * 4);
  float*    p2    = (float*)   alloc((size_t)NB2 * 128 * 4);

  {
    int total = 3 * 128 * 192 + 2 * 48 * 64 + 64 * 192;
    prep_kernel<<<(total + 255) / 256, 256, 0, stream>>>(W_full, W_scale, W_fuse,
                                                         WT1, WTs, WTf);
  }
  for (int i = 0; i < 3; ++i) {
    if (i == 0) {
      long total = (long)ROWS1 * NF_PAD;
      nf0_kernel<<<(int)((total + 255) / 256), 256, 0, stream>>>(nbr, nf16);
    } else {
      nfgemm_kernel<<<600, 128, 0, stream>>>(nbr, WTs + (size_t)(i - 1) * 48 * 64,
                                             b_scale + (i - 1) * FFEA, nf16);
    }
    pass1_kernel<<<NB1, 128, 0, stream>>>(atom, idx, nf16,
                                          WT1 + (size_t)i * 128 * 192,
                                          b_full + i * 128, z16, p1);
    reduce1_kernel<<<1, 128, 0, stream>>>(p1, g1 + i * 128, bt1 + i * 128, bn1);
    pass2_kernel<<<NB2, 256, 0, stream>>>(z16, bn1, summed, p2);
    reduce2_kernel<<<1, 64, 0, stream>>>(p2, g2 + i * 64, bt2 + i * 64, bn2);
    {
      long total = (long)NATOMS * 64;
      pass3_kernel<<<(int)((total + 255) / 256), 256, 0, stream>>>(atom, summed, bn2,
                                                                   cat16, i);
    }
  }
  fuse_kernel<<<400, 128, 0, stream>>>(cat16, WTf, b_fuse, out);

  (void)in_sizes; (void)n_in; (void)out_size; (void)ws_size;
}
